// TriangleAttentionEndingNode_63531156242733
// MI455X (gfx1250) — matrix operand order repaired and checked by reading, not compiled, not run on hardware
//
#include <hip/hip_runtime.h>
#include <hip/hip_bf16.h>
#include <cstdint>

typedef __attribute__((ext_vector_type(16))) __bf16 v16bf;
typedef __attribute__((ext_vector_type(8)))  float  v8f;

#define NRES  256
#define DPAIR 128
#define NHEAD 4
#define CHEAD 32
#define NROWS (NRES*NRES)   // 65536
#define EQKV  384
#define ECAT  512

static __device__ __forceinline__ float wred_sum(float x){
#pragma unroll
  for (int o = 16; o > 0; o >>= 1) x += __shfl_xor(x, o, 32);
  return x;
}
static __device__ __forceinline__ float wred_max(float x){
#pragma unroll
  for (int o = 16; o > 0; o >>= 1) x = fmaxf(x, __shfl_xor(x, o, 32));
  return x;
}

union frag16 { uint4 u[2]; v16bf v; __bf16 h[16]; };

// A-fragment: lane holds row (lane&15), K elems at [c0..c0+7] then [c1..c1+7]
static __device__ __forceinline__ v16bf load_a_frag(const __bf16* rowp, int c0, int c1){
  frag16 f;
  f.u[0] = *reinterpret_cast<const uint4*>(rowp + c0);
  f.u[1] = *reinterpret_cast<const uint4*>(rowp + c1);
  return f.v;
}
// B-fragment: lane holds col (lane&15), 16 contiguous K values from p
static __device__ __forceinline__ v16bf load_b_frag(const __bf16* p){
  frag16 f;
  const uint4* q = reinterpret_cast<const uint4*>(p);
  f.u[0] = q[0]; f.u[1] = q[1];
  return f.v;
}
static __device__ __forceinline__ v8f wmma_bf16(v16bf a, v16bf b, v8f c){
  return __builtin_amdgcn_wmma_f32_16x16x32_bf16(false, a, false, b, (short)0, c, false, false);
}
static __device__ __forceinline__ v8f zero8(){
  v8f z;
#pragma unroll
  for (int i = 0; i < 8; i++) z[i] = 0.f;
  return z;
}

// LDS byte offset of a __shared__ object (generic -> addrspace(3) -> int)
static __device__ __forceinline__ unsigned lds_offset(const void* p){
  return (unsigned)(unsigned long long)(__attribute__((address_space(3))) const void*)p;
}
// Async memory->LDS copy, 16B per lane, tracked on ASYNCcnt (CDNA5 §15.18.3 op 98)
static __device__ __forceinline__ void async_copy_b128(unsigned lds_addr, const void* gptr){
  asm volatile("global_load_async_to_lds_b128 %0, %1, off"
               :: "v"(lds_addr), "v"(gptr) : "memory");
}
static __device__ __forceinline__ void wait_asynccnt0(){
  asm volatile("s_wait_asynccnt 0x0" ::: "memory");
}

// ---------------- Kernel 0: weight bf16 conversion ----------------
__global__ __launch_bounds__(256) void k_wconv(
    const float* __restrict__ wqkv, const float* __restrict__ wg,
    const float* __restrict__ wo, __bf16* __restrict__ wcat,
    __bf16* __restrict__ wob)
{
  int idx = blockIdx.x * blockDim.x + threadIdx.x;
  if (idx < ECAT * DPAIR) {
    int e = idx / DPAIR, d = idx % DPAIR;
    float v = (e < EQKV) ? wqkv[e * DPAIR + d] : wg[(e - EQKV) * DPAIR + d];
    wcat[idx] = (__bf16)v;
  }
  if (idx < DPAIR * DPAIR) wob[idx] = (__bf16)wo[idx];
}

// ------- Kernel 1: transpose + LayerNorm + pair-bias (1 wave / row) -------
__global__ __launch_bounds__(256) void k_ln_bias(
    const float* __restrict__ Zraw, const float* __restrict__ gamma,
    const float* __restrict__ beta, const float* __restrict__ wb,
    __bf16* __restrict__ zn, float* __restrict__ biasbuf)
{
  int wave = threadIdx.x >> 5, lane = threadIdx.x & 31;
  int r = blockIdx.x * 8 + wave;           // r = i*256 + j  (Zn row)
  int i = r >> 8, j = r & 255;
  const float* src = Zraw + ((size_t)j * NRES + i) * DPAIR;  // transpose read
  float4 xl = *reinterpret_cast<const float4*>(src + lane * 4);
  float x[4] = {xl.x, xl.y, xl.z, xl.w};
  float mu = wred_sum(x[0] + x[1] + x[2] + x[3]) * (1.0f / DPAIR);
  float vs = 0.f;
#pragma unroll
  for (int t = 0; t < 4; t++) { float d = x[t] - mu; vs += d * d; }
  float rstd = rsqrtf(wred_sum(vs) * (1.0f / DPAIR) + 1e-5f);
  float zv[4];
  union { uint2 u; __bf16 h[4]; } pk;
#pragma unroll
  for (int t = 0; t < 4; t++) {
    int ch = lane * 4 + t;
    zv[t] = (x[t] - mu) * rstd * gamma[ch] + beta[ch];
    pk.h[t] = (__bf16)zv[t];
  }
  *reinterpret_cast<uint2*>(zn + (size_t)r * DPAIR + lane * 4) = pk.u;
#pragma unroll
  for (int h = 0; h < NHEAD; h++) {
    float p = 0.f;
#pragma unroll
    for (int t = 0; t < 4; t++) p += zv[t] * wb[h * DPAIR + lane * 4 + t];
    p = wred_sum(p);
    if (lane == 0) biasbuf[(size_t)h * NROWS + r] = p;
  }
}

// ------- Kernel 2: fused QKV+gate projection, WMMA [65536x128]x[128x512] -------
// Weight tile (64 cols x 128 K = 16 KB) is staged memory->LDS via async copies
// (ASYNCcnt path), then all 8 waves consume B-fragments from LDS.
__global__ __launch_bounds__(256) void k_proj(
    const __bf16* __restrict__ zn, const __bf16* __restrict__ wcat,
    const float* __restrict__ bg,
    __bf16* __restrict__ qb, __bf16* __restrict__ kb,
    __bf16* __restrict__ vtb, float* __restrict__ gb)
{
  __shared__ __bf16 sB[64 * DPAIR];   // 16 KB weight tile
  int wave = threadIdx.x >> 5, lane = threadIdx.x & 31;
  int m0 = blockIdx.x * 128 + wave * 16;
  int nbase = blockIdx.y * 64;
  int lrow = lane & 15, lhalf = lane >> 4;

  // async stage: 1024 x 16B chunks, 4 per thread
  {
    unsigned base = lds_offset(sB);
    const char* gsrc = (const char*)(wcat + (size_t)nbase * DPAIR);
#pragma unroll
    for (int t = 0; t < 4; t++) {
      unsigned idx = threadIdx.x + t * 256;
      async_copy_b128(base + idx * 16, gsrc + (size_t)idx * 16);
    }
    wait_asynccnt0();
    __syncthreads();
  }

  const __bf16* arow = zn + (size_t)(m0 + lrow) * DPAIR;
  v8f acc[4];
#pragma unroll
  for (int nt = 0; nt < 4; nt++) acc[nt] = zero8();
#pragma unroll
  for (int kk = 0; kk < 4; kk++) {
    int k0 = kk * 32;
    v16bf a = load_a_frag(arow, k0 + lhalf * 16, k0 + lhalf * 16 + 8);
#pragma unroll
    for (int nt = 0; nt < 4; nt++) {
      v16bf b = load_b_frag(sB + (size_t)(nt * 16 + lrow) * DPAIR + k0 + lhalf * 16);
      acc[nt] = wmma_bf16(a, b, acc[nt]);
    }
  }
  const float QSCALE = 0.17677669529663687f;  // C^-0.5
#pragma unroll
  for (int nt = 0; nt < 4; nt++) {
    int e = nbase + nt * 16 + lrow;
#pragma unroll
    for (int vv = 0; vv < 8; vv++) {
      int row = m0 + vv + lhalf * 8;
      int i = row >> 8, j = row & 255;
      float val = acc[nt][vv];
      if (e < 128) {                         // q, pre-scaled
        int h = e >> 5, c = e & 31;
        qb[((size_t)(i * NHEAD + h) * NRES + j) * CHEAD + c] = (__bf16)(val * QSCALE);
      } else if (e < 256) {                  // k
        int e2 = e - 128, h = e2 >> 5, c = e2 & 31;
        kb[((size_t)(i * NHEAD + h) * NRES + j) * CHEAD + c] = (__bf16)val;
      } else if (e < 384) {                  // v, transposed [c][j]
        int e2 = e - 256, h = e2 >> 5, c = e2 & 31;
        vtb[((size_t)(i * NHEAD + h) * CHEAD + c) * NRES + j] = (__bf16)val;
      } else {                               // gate, sigmoid fused
        int e2 = e - 384;
        gb[(size_t)row * DPAIR + e2] = 1.f / (1.f + __expf(-(val + bg[e2])));
      }
    }
  }
}

// ------- Kernel 3: attention per (i,h,q-tile): QK^T + softmax + PV -------
__global__ __launch_bounds__(64) void k_attn(
    const __bf16* __restrict__ qb, const __bf16* __restrict__ kb,
    const __bf16* __restrict__ vtb, const float* __restrict__ biasbuf,
    const float* __restrict__ zmask, float* __restrict__ ob)
{
  __shared__ float  lds_l[2][16][NRES];   // 32 KB
  __shared__ __bf16 lds_p[2][16][NRES];   // 16 KB
  int wave = threadIdx.x >> 5, lane = threadIdx.x & 31;
  int ih = blockIdx.x >> 3;               // i*NHEAD + h
  int i = ih >> 2, h = ih & 3;
  int j0 = ((blockIdx.x & 7) * 2 + wave) * 16;
  int lrow = lane & 15, lhalf = lane >> 4;
  const __bf16* qbase = qb + (size_t)ih * NRES * CHEAD;
  const __bf16* kbase = kb + (size_t)ih * NRES * CHEAD;
  const __bf16* vbase = vtb + (size_t)ih * CHEAD * NRES;

  // QK^T: K = C = 32 (one WMMA per 16-col tile)
  v16bf aq = load_a_frag(qbase + (size_t)(j0 + lrow) * CHEAD, lhalf * 16, lhalf * 16 + 8);
#pragma unroll
  for (int kt = 0; kt < 16; kt++) {
    v16bf bk = load_b_frag(kbase + (size_t)(kt * 16 + lrow) * CHEAD + lhalf * 16);
    v8f acc = wmma_bf16(aq, bk, zero8());
    int colg = kt * 16 + lrow;
    float mterm = 1e9f * (zmask[(size_t)colg * NRES + i] - 1.0f);
#pragma unroll
    for (int vv = 0; vv < 8; vv++) {
      int qr = j0 + vv + lhalf * 8;
      lds_l[wave][vv + lhalf * 8][colg] =
          acc[vv] + biasbuf[(size_t)h * NROWS + (size_t)qr * NRES + colg] + mterm;
    }
  }
  // softmax over 256 cols, wave-local (8 elems/lane/row)
#pragma unroll 1
  for (int rr = 0; rr < 16; rr++) {
    const float* rp = &lds_l[wave][rr][lane * 8];
    float x[8];
#pragma unroll
    for (int t = 0; t < 8; t++) x[t] = rp[t];
    float m = x[0];
#pragma unroll
    for (int t = 1; t < 8; t++) m = fmaxf(m, x[t]);
    m = wred_max(m);
    float s = 0.f;
#pragma unroll
    for (int t = 0; t < 8; t++) { x[t] = __expf(x[t] - m); s += x[t]; }
    float inv = 1.f / wred_sum(s);
#pragma unroll
    for (int t = 0; t < 8; t++) lds_p[wave][rr][lane * 8 + t] = (__bf16)(x[t] * inv);
  }
  // o = P[16x256] @ V[256x32]
  v8f oacc[2]; oacc[0] = zero8(); oacc[1] = zero8();
#pragma unroll
  for (int ks = 0; ks < 8; ks++) {
    int k0 = ks * 32;
    v16bf ap = load_a_frag(&lds_p[wave][lrow][0], k0 + lhalf * 16, k0 + lhalf * 16 + 8);
#pragma unroll
    for (int nt = 0; nt < 2; nt++) {
      int c = nt * 16 + lrow;
      v16bf bv = load_b_frag(vbase + (size_t)c * NRES + k0 + lhalf * 16);
      oacc[nt] = wmma_bf16(ap, bv, oacc[nt]);
    }
  }
#pragma unroll
  for (int nt = 0; nt < 2; nt++)
#pragma unroll
    for (int vv = 0; vv < 8; vv++) {
      int j = j0 + vv + lhalf * 8;
      int c = nt * 16 + lrow;
      ob[((size_t)i * NRES + j) * DPAIR + h * CHEAD + c] = oacc[nt][vv];
    }
}

// ------- Kernel 4: (g*o) @ w_o^T + out_bias + residual, transposed store -------
__global__ __launch_bounds__(256) void k_out(
    const float* __restrict__ ob, const float* __restrict__ gb,
    const __bf16* __restrict__ wob, const float* __restrict__ obias,
    const float* __restrict__ Zraw, float* __restrict__ out)
{
  int wave = threadIdx.x >> 5, lane = threadIdx.x & 31;
  int m0 = blockIdx.x * 128 + wave * 16;
  int lrow = lane & 15, lhalf = lane >> 4;
  const float* orow = ob + (size_t)(m0 + lrow) * DPAIR;
  const float* grow = gb + (size_t)(m0 + lrow) * DPAIR;

  // prefetch residual rows consumed only in the epilogue (global_prefetch_b8)
  {
    int row = m0 + lrow;
    int i = row >> 8, j = row & 255;
    __builtin_prefetch(Zraw + ((size_t)j * NRES + i) * DPAIR, 0, 3);
  }

  v8f acc[8];
#pragma unroll
  for (int nt = 0; nt < 8; nt++) acc[nt] = zero8();
#pragma unroll
  for (int kk = 0; kk < 4; kk++) {
    int k0 = kk * 32;
    int c0 = k0 + lhalf * 16, c1 = k0 + lhalf * 16 + 8;
    frag16 af;
#pragma unroll
    for (int t = 0; t < 8; t++) af.h[t]     = (__bf16)(orow[c0 + t] * grow[c0 + t]);
#pragma unroll
    for (int t = 0; t < 8; t++) af.h[8 + t] = (__bf16)(orow[c1 + t] * grow[c1 + t]);
#pragma unroll
    for (int nt = 0; nt < 8; nt++) {
      int d = nt * 16 + lrow;
      v16bf b = load_b_frag(wob + (size_t)d * DPAIR + k0 + lhalf * 16);
      acc[nt] = wmma_bf16(af.v, b, acc[nt]);
    }
  }
#pragma unroll
  for (int nt = 0; nt < 8; nt++)
#pragma unroll
    for (int vv = 0; vv < 8; vv++) {
      int row = m0 + vv + lhalf * 8;
      int i = row >> 8, j = row & 255;
      int d = nt * 16 + lrow;
      size_t oi = ((size_t)j * NRES + i) * DPAIR + d;   // transpose back
      out[oi] = acc[nt][vv] + obias[d] + Zraw[oi];      // + residual
    }
}

extern "C" void kernel_launch(void* const* d_in, const int* in_sizes, int n_in,
                              void* d_out, int out_size, void* d_ws, size_t ws_size,
                              hipStream_t stream)
{
  (void)in_sizes; (void)n_in; (void)out_size; (void)ws_size;
  const float* Zraw  = (const float*)d_in[0];
  const float* Zmask = (const float*)d_in[1];
  const float* gamma = (const float*)d_in[2];
  const float* beta  = (const float*)d_in[3];
  const float* wb    = (const float*)d_in[4];
  const float* wqkv  = (const float*)d_in[5];
  const float* wg    = (const float*)d_in[6];
  const float* bg    = (const float*)d_in[7];
  const float* wo    = (const float*)d_in[8];
  const float* obias = (const float*)d_in[9];
  float* out = (float*)d_out;

  char* ws = (char*)d_ws;
  size_t off = 0;
  auto alloc = [&](size_t bytes) -> void* {
    void* p = ws + off;
    off += (bytes + 255) & ~(size_t)255;
    return p;
  };
  __bf16* zn    = (__bf16*)alloc((size_t)NROWS * DPAIR * 2);
  __bf16* qb    = (__bf16*)alloc((size_t)NROWS * DPAIR * 2);
  __bf16* kb    = (__bf16*)alloc((size_t)NROWS * DPAIR * 2);
  __bf16* vtb   = (__bf16*)alloc((size_t)NROWS * DPAIR * 2);
  float*  biasb = (float*) alloc((size_t)NHEAD * NROWS * 4);
  float*  gbuf  = (float*) alloc((size_t)NROWS * DPAIR * 4);
  float*  obuf  = (float*) alloc((size_t)NROWS * DPAIR * 4);
  __bf16* wcat  = (__bf16*)alloc((size_t)ECAT * DPAIR * 2);
  __bf16* wobb  = (__bf16*)alloc((size_t)DPAIR * DPAIR * 2);

  k_wconv  <<<256, 256, 0, stream>>>(wqkv, wg, wo, wcat, wobb);
  k_ln_bias<<<NROWS / 8, 256, 0, stream>>>(Zraw, gamma, beta, wb, zn, biasb);
  k_proj   <<<dim3(NROWS / 128, ECAT / 64), 256, 0, stream>>>(zn, wcat, bg, qb, kb, vtb, gbuf);
  k_attn   <<<NRES * NHEAD * 8, 64, 0, stream>>>(qb, kb, vtb, biasb, Zmask, obuf);
  k_out    <<<NROWS / 128, 256, 0, stream>>>(obuf, gbuf, wobb, obias, Zraw, out);
}
